// DimeNetDiabat_22256520528856
// MI455X (gfx1250) — compile-verified, matrix operand order checked by reading, not run on hardware
//
#include <hip/hip_runtime.h>
#include <math.h>

// ---------------------------------------------------------------------------
// DimeNet-Diabat forward for MI455X (gfx1250, wave32, WMMA)
// All dense [*,128]x[128,128]/[384,128]/[128,336] GEMMs go through
// v_wmma_f32_16x16x32_f16 (f16 operands, f32 accumulation), with the
// B (weight) operand staged in LDS (bank-conflict-padded) and reused by
// all 8 waves of the block.  GEMM waves compute 32x64 tiles so each LDS
// B fragment feeds two WMMAs.
// ---------------------------------------------------------------------------

typedef __attribute__((ext_vector_type(16))) _Float16 v16h;
typedef __attribute__((ext_vector_type(8)))  float    v8f;

#define WMMA_F16F32(a, b, c) \
  __builtin_amdgcn_wmma_f32_16x16x32_f16(false, (a), false, (b), (short)0, (c), false, false)

#define PI_F 3.14159265358979f
#define LDS_STRIDE 136   // 128 halfs + 8 pad -> 272B row stride (4-bank skew)

// spherical Bessel roots j_l, l=0..6, first 6 roots each (precomputed constant)
__constant__ float c_roots[7][6] = {
  { 3.14159265f,  6.28318531f,  9.42477796f, 12.56637061f, 15.70796327f, 18.84955592f},
  { 4.49340946f,  7.72525184f, 10.90412166f, 14.06619391f, 17.22075527f, 20.37130296f},
  { 5.76345920f,  9.09501133f, 12.32294097f, 15.51460301f, 18.68903636f, 21.85387450f},
  { 6.98793200f, 10.41711855f, 13.69802315f, 16.92362129f, 20.12180617f, 23.30424718f},
  { 8.18256145f, 11.70490715f, 15.03966471f, 18.30125596f, 21.52541806f, 24.72756634f},
  { 9.35581211f, 12.96653017f, 16.35470964f, 19.65315219f, 22.90455104f, 26.12666830f},
  {10.51283540f, 14.20739215f, 17.64797495f, 20.98346311f, 24.26276804f, 27.50786837f}
};

__device__ __forceinline__ float silu_f(float x) { return x / (1.f + __expf(-x)); }

// envelope for p=6: 1/x - 28 x^5 + 48 x^6 - 21 x^7
__device__ __forceinline__ float envelope6(float x) {
  float x2 = x * x;
  float x5 = x2 * x2 * x;
  return 1.f / x + (-28.f) * x5 + 48.f * x5 * x + (-21.f) * x5 * x2;
}

__device__ __forceinline__ float sphjl(int l, float x) {
  float j0 = __sinf(x) / x;
  if (l == 0) return j0;
  float j1 = __sinf(x) / (x * x) - __cosf(x) / x;
  float jp = j0, jc = j1;
  for (int ll = 1; ll < l; ++ll) {
    float jn = (2.f * ll + 1.f) / x * jc - jp;
    jp = jc; jc = jn;
  }
  return jc;
}

// ---------------------------------------------------------------------------
// Geometry kernels
// ---------------------------------------------------------------------------
__global__ void k_edge_geom(const float* __restrict__ xyz, const int* __restrict__ nbr,
                            float* __restrict__ dE, float* __restrict__ erbf, int E) {
  int e = blockIdx.x * blockDim.x + threadIdx.x;
  if (e >= E) return;
  int n0 = nbr[2 * e], n1 = nbr[2 * e + 1];
  float dx = xyz[3 * n0] - xyz[3 * n1];
  float dy = xyz[3 * n0 + 1] - xyz[3 * n1 + 1];
  float dz = xyz[3 * n0 + 2] - xyz[3 * n1 + 2];
  float d = sqrtf(dx * dx + dy * dy + dz * dz + 1e-12f);
  dE[e] = d;
  float x = d * 0.2f;               // d / CUTOFF
  float env = envelope6(x);
#pragma unroll
  for (int r = 0; r < 6; ++r)
    erbf[e * 6 + r] = env * __sinf((r + 1) * PI_F * x);
}

__global__ void k_sbf(const float* __restrict__ xyz, const int* __restrict__ ang,
                      const int* __restrict__ kj, const float* __restrict__ dE,
                      float* __restrict__ sbf, int T) {
  int t = blockIdx.x * blockDim.x + threadIdx.x;
  if (t >= T) return;
  int a0 = ang[3 * t], a1 = ang[3 * t + 1], a2 = ang[3 * t + 2];
  float jx = xyz[3 * a0]     - xyz[3 * a1];
  float jy = xyz[3 * a0 + 1] - xyz[3 * a1 + 1];
  float jz = xyz[3 * a0 + 2] - xyz[3 * a1 + 2];
  float kx = xyz[3 * a2]     - xyz[3 * a1];
  float ky = xyz[3 * a2 + 1] - xyz[3 * a1 + 1];
  float kz = xyz[3 * a2 + 2] - xyz[3 * a1 + 2];
  float xx = jx * kx + jy * ky + jz * kz;
  float cx = jy * kz - jz * ky;
  float cy = jz * kx - jx * kz;
  float cz = jx * ky - jy * kx;
  float yy = sqrtf(cx * cx + cy * cy + cz * cz + 1e-12f);
  float alpha = atan2f(yy, xx);
  float x = dE[kj[t]] * 0.2f;
  float env = envelope6(x);
  float ca = __cosf(alpha);
  float P[7];
  P[0] = 1.f; P[1] = ca;
#pragma unroll
  for (int l = 1; l < 6; ++l)
    P[l + 1] = ((2.f * l + 1.f) * ca * P[l] - (float)l * P[l - 1]) / (float)(l + 1);
  float* row = sbf + (size_t)t * 42;
#pragma unroll
  for (int l = 0; l < 7; ++l) {
    float Yl = sqrtf((2.f * l + 1.f) * 0.07957747155f) * P[l];
#pragma unroll
    for (int n = 0; n < 6; ++n) {
      float arg = x * c_roots[l][n];
      if (arg < 1e-6f) arg = 1e-6f;
      row[l * 6 + n] = env * sphjl(l, arg) * Yl;
    }
  }
}

// ---------------------------------------------------------------------------
// Weight conversion: f32 [K,N] -> f16 transposed [N,K] (B-fragment friendly)
// ---------------------------------------------------------------------------
__global__ void k_w_tr(const float* __restrict__ W, _Float16* __restrict__ Wt, int K, int N) {
  int idx = blockIdx.x * blockDim.x + threadIdx.x;
  if (idx >= K * N) return;
  int k = idx / N, n = idx % N;
  Wt[(size_t)n * K + k] = (_Float16)W[idx];
}

// bilinear W [42,128,8] -> Wt [(s*8+b)=336, 128]
__global__ void k_bilin_tr(const float* __restrict__ W, _Float16* __restrict__ Wt) {
  int idx = blockIdx.x * blockDim.x + threadIdx.x;
  if (idx >= 42 * 128 * 8) return;
  int s = idx / 1024;
  int r = idx % 1024;
  int d = r >> 3, b = r & 7;
  Wt[(size_t)(s * 8 + b) * 128 + d] = (_Float16)W[idx];
}

__global__ void k_cast_h(const float* __restrict__ in, _Float16* __restrict__ out, int n) {
  int i = blockIdx.x * blockDim.x + threadIdx.x;
  if (i < n) out[i] = (_Float16)in[i];
}

// ---------------------------------------------------------------------------
// Generic WMMA GEMM: out = epi( A_f16[M,K] @ Wt_f16[128,K]^T )
// B staged in LDS per 128-wide K chunk (34KB, padded stride).  N fixed 128.
// block = 256 threads (8 waves) covering a 128x128 output tile;
// wave (rg, cg) computes 32 rows x 64 cols = 2 M-tiles x 4 N-tiles, so each
// LDS B fragment is consumed by two WMMAs.
// ---------------------------------------------------------------------------
__device__ __forceinline__ void gemm_epi(v8f acc, int nt, long rowB, int l16, int hsel,
                                         const float* bias, const float* mul,
                                         const float* resid, float* outF,
                                         _Float16* outH, int M, int actSilu) {
  int col = nt * 16 + l16;
  float bv = bias ? bias[col] : 0.f;
#pragma unroll
  for (int v = 0; v < 8; ++v) {
    long r = rowB + v + hsel * 8;
    if (r < M) {
      float y = acc[v] + bv;
      if (actSilu) y = silu_f(y);
      long o = r * 128 + col;
      if (mul)   y *= mul[o];
      if (resid) y += resid[o];
      if (outF)  outF[o] = y;
      if (outH)  outH[o] = (_Float16)y;
    }
  }
}

__global__ __launch_bounds__(256, 1)
void k_gemm128(const _Float16* __restrict__ A, const _Float16* __restrict__ WT,
               const float* __restrict__ bias, const float* __restrict__ mul,
               const float* __restrict__ resid, float* __restrict__ outF,
               _Float16* __restrict__ outH, int M, int K, int actSilu) {
  __shared__ _Float16 ldsB[128 * LDS_STRIDE];      // 34KB, bank-skewed rows
  const int tid  = threadIdx.x;
  const int lane = tid & 31;
  const int wave = tid >> 5;
  const int rg   = wave >> 1;                      // row group: 0..3 (32 rows)
  const int cg   = wave & 1;                       // col group: 0..1 (64 cols)
  const int l16  = lane & 15;
  const int hsel = lane >> 4;
  const long rowBase = (long)blockIdx.x * 128 + rg * 32;
  const bool active = rowBase < M;
  long aRow0 = rowBase + l16;       if (aRow0 >= M) aRow0 = M - 1;
  long aRow1 = rowBase + 16 + l16;  if (aRow1 >= M) aRow1 = M - 1;
  if (!active) { aRow0 = 0; aRow1 = 0; }
  const _Float16* aPtr0 = A + aRow0 * (long)K + hsel * 16;
  const _Float16* aPtr1 = A + aRow1 * (long)K + hsel * 16;
  __builtin_prefetch(aPtr0, 0, 0);                 // global_prefetch_b8
  v8f c00 = {}, c01 = {}, c02 = {}, c03 = {};
  v8f c10 = {}, c11 = {}, c12 = {}, c13 = {};
  for (int kc = 0; kc < K; kc += 128) {
    __syncthreads();                               // prior chunk reads complete
    // cooperative stage of WT[0:128, kc:kc+128] -> LDS (2048 x 16B)
    for (int i = tid; i < 2048; i += 256) {
      int row = i >> 4;
      int co  = (i & 15) * 8;
      *(float4*)(&ldsB[row * LDS_STRIDE + co]) =
          *(const float4*)(WT + (size_t)row * K + kc + co);
    }
    __syncthreads();
    if (active) {
      const _Float16* bb = &ldsB[(cg * 64 + l16) * LDS_STRIDE + hsel * 16];
#pragma unroll
      for (int kk = 0; kk < 128; kk += 32) {
        v16h a0 = *(const v16h*)(aPtr0 + kc + kk);
        v16h a1 = *(const v16h*)(aPtr1 + kc + kk);
        v16h b0 = *(const v16h*)(bb + 0 * 16 * LDS_STRIDE + kk);
        c00 = WMMA_F16F32(a0, b0, c00);
        c10 = WMMA_F16F32(a1, b0, c10);
        v16h b1 = *(const v16h*)(bb + 1 * 16 * LDS_STRIDE + kk);
        c01 = WMMA_F16F32(a0, b1, c01);
        c11 = WMMA_F16F32(a1, b1, c11);
        v16h b2 = *(const v16h*)(bb + 2 * 16 * LDS_STRIDE + kk);
        c02 = WMMA_F16F32(a0, b2, c02);
        c12 = WMMA_F16F32(a1, b2, c12);
        v16h b3 = *(const v16h*)(bb + 3 * 16 * LDS_STRIDE + kk);
        c03 = WMMA_F16F32(a0, b3, c03);
        c13 = WMMA_F16F32(a1, b3, c13);
      }
    }
  }
  if (!active) return;
  const int ntB = cg * 4;
  gemm_epi(c00, ntB + 0, rowBase,      l16, hsel, bias, mul, resid, outF, outH, M, actSilu);
  gemm_epi(c01, ntB + 1, rowBase,      l16, hsel, bias, mul, resid, outF, outH, M, actSilu);
  gemm_epi(c02, ntB + 2, rowBase,      l16, hsel, bias, mul, resid, outF, outH, M, actSilu);
  gemm_epi(c03, ntB + 3, rowBase,      l16, hsel, bias, mul, resid, outF, outH, M, actSilu);
  gemm_epi(c10, ntB + 0, rowBase + 16, l16, hsel, bias, mul, resid, outF, outH, M, actSilu);
  gemm_epi(c11, ntB + 1, rowBase + 16, l16, hsel, bias, mul, resid, outF, outH, M, actSilu);
  gemm_epi(c12, ntB + 2, rowBase + 16, l16, hsel, bias, mul, resid, outF, outH, M, actSilu);
  gemm_epi(c13, ntB + 3, rowBase + 16, l16, hsel, bias, mul, resid, outF, outH, M, actSilu);
}

// ---------------------------------------------------------------------------
// Bilinear + segment-sum kernel (the dominant 34 GF/layer GEMM, fused):
//   Y[t, s*8+b] = x_kj[kj_idx[t], :] @ Wt[s*8+b, :]      (WMMA, 21 N-tiles)
//   tb[t, b]   += sbf[t, s] * Y[t, s*8+b]                (LDS ds_add_f32)
//   agg[ji_idx[t], b] += tb[t, b]                        (global_atomic_add_f32)
// B staged in LDS (3 chunks of 112 rows); per-wave sbf rows staged in LDS.
// Wave = 16 triplets; A-fragments gathered through kj_idx at load time.
// ---------------------------------------------------------------------------
__global__ __launch_bounds__(256, 1)
void k_bilin(const _Float16* __restrict__ Xkj, const _Float16* __restrict__ WT,
             const float* __restrict__ sbf, const int* __restrict__ kj,
             const int* __restrict__ ji, float* __restrict__ agg, int T) {
  __shared__ _Float16 ldsB[112 * LDS_STRIDE];      // 30.4KB
  __shared__ float tb[8][16][8];                   // 4KB
  __shared__ float sbfL[8][16][42];                // 21.5KB
  const int tid  = threadIdx.x;
  const int lane = tid & 31;
  const int wave = tid >> 5;
  const int l16  = lane & 15;
  const int hsel = lane >> 4;
  const long tBase = (long)blockIdx.x * 128 + wave * 16;
  const bool active = tBase < T;
  for (int i = lane; i < 128; i += 32) tb[wave][i >> 3][i & 7] = 0.f;
  // stage this wave's sbf rows: 16 x 42
  for (int i = lane; i < 672; i += 32) {
    int r = i / 42, s = i % 42;
    long t = tBase + r;
    sbfL[wave][r][s] = (t < T) ? sbf[t * 42 + s] : 0.f;
  }
  long tr = tBase + l16;
  if (tr >= T) tr = T - 1;
  if (!active) tr = 0;
  long e = kj[tr];
  const _Float16* ap = Xkj + e * 128 + hsel * 16;
  __builtin_prefetch(ap, 0, 0);
  v16h af0 = *(const v16h*)(ap);
  v16h af1 = *(const v16h*)(ap + 32);
  v16h af2 = *(const v16h*)(ap + 64);
  v16h af3 = *(const v16h*)(ap + 96);
  for (int ntc = 0; ntc < 21; ntc += 7) {
    __syncthreads();                               // prior chunk reads complete
    // stage WT rows [ntc*16, ntc*16+112) -> LDS (1792 x 16B)
    for (int i = tid; i < 1792; i += 256) {
      int row = i >> 4;
      int co  = (i & 15) * 8;
      *(float4*)(&ldsB[row * LDS_STRIDE + co]) =
          *(const float4*)(WT + (size_t)(ntc * 16 + row) * 128 + co);
    }
    __syncthreads();
    if (active) {
      const _Float16* bb = &ldsB[l16 * LDS_STRIDE + hsel * 16];
      for (int n2 = 0; n2 < 7; ++n2) {
        const _Float16* bp = bb + n2 * 16 * LDS_STRIDE;
        v8f c = {};
        c = WMMA_F16F32(af0, *(const v16h*)(bp),      c);
        c = WMMA_F16F32(af1, *(const v16h*)(bp + 32), c);
        c = WMMA_F16F32(af2, *(const v16h*)(bp + 64), c);
        c = WMMA_F16F32(af3, *(const v16h*)(bp + 96), c);
        int col = (ntc + n2) * 16 + l16;           // 0..335 -> (s, b)
        int s = col >> 3, b = col & 7;
#pragma unroll
        for (int v = 0; v < 8; ++v) {
          int rl = v + hsel * 8;
          long t = tBase + rl;
          if (t < T) {
            float sv = sbfL[wave][rl][s];
            atomicAdd(&tb[wave][rl][b], c[v] * sv);   // ds_add_f32
          }
        }
      }
    }
  }
  __syncthreads();
#pragma unroll
  for (int j2 = 0; j2 < 4; ++j2) {
    int idx = lane * 4 + j2;
    int rl = idx >> 3, b = idx & 7;
    long t = tBase + rl;
    if (t < T) atomicAdd(&agg[(size_t)ji[t] * 8 + b], tb[wave][rl][b]);
  }
}

// ---------------------------------------------------------------------------
// Small VALU kernels
// ---------------------------------------------------------------------------
// Build embedding input [E,384] f16 = [h_j | h_i | silu(e_rbf@W+b)]
__global__ void k_build_emb(const int* __restrict__ nbr, const int* __restrict__ z,
                            const float* __restrict__ emb_table,
                            const float* __restrict__ erbf,
                            const float* __restrict__ rbfW, const float* __restrict__ rbfB,
                            _Float16* __restrict__ Aemb, int E) {
  int idx = blockIdx.x * blockDim.x + threadIdx.x;
  if (idx >= E * 128) return;
  int e = idx >> 7, c = idx & 127;
  int j = nbr[2 * e], i = nbr[2 * e + 1];
  float hj = emb_table[z[j] * 128 + c];
  float hi = emb_table[z[i] * 128 + c];
  float s = rbfB[c];
#pragma unroll
  for (int r = 0; r < 6; ++r) s += erbf[e * 6 + r] * rbfW[r * 128 + c];
  s = silu_f(s);
  _Float16* row = Aemb + (size_t)e * 384;
  row[c] = (_Float16)hj;
  row[128 + c] = (_Float16)hi;
  row[256 + c] = (_Float16)s;
}

// mul[e,c] = sum_r e_rbf[e,r] * W[r,c]   (rbf projection for x_kj)
__global__ void k_rbfproj(const float* __restrict__ erbf, const float* __restrict__ W,
                          float* __restrict__ out, int E) {
  int idx = blockIdx.x * blockDim.x + threadIdx.x;
  if (idx >= E * 128) return;
  int e = idx >> 7, c = idx & 127;
  float g = 0.f;
#pragma unroll
  for (int r = 0; r < 6; ++r) g += erbf[e * 6 + r] * W[r * 128 + c];
  out[idx] = g;
}

// output block edge part: t_atom[nbr_i, c] += (e_rbf@W)[e,c] * m[e,c]
__global__ void k_out_edge(const float* __restrict__ erbf, const float* __restrict__ W,
                           const float* __restrict__ m, const int* __restrict__ nbr,
                           float* __restrict__ tacc, int E) {
  int idx = blockIdx.x * blockDim.x + threadIdx.x;
  if (idx >= E * 128) return;
  int e = idx >> 7, c = idx & 127;
  float g = 0.f;
#pragma unroll
  for (int r = 0; r < 6; ++r) g += erbf[e * 6 + r] * W[r * 128 + c];
  atomicAdd(&tacc[(size_t)nbr[2 * e + 1] * 128 + c], g * m[idx]); // global_atomic_add_f32
}

// x = x_ji + agg @ up_W   -> f32 + f16
__global__ void k_up(const float* __restrict__ xji, const float* __restrict__ agg,
                     const float* __restrict__ upW, float* __restrict__ xF,
                     _Float16* __restrict__ xH, int E) {
  int idx = blockIdx.x * blockDim.x + threadIdx.x;
  if (idx >= E * 128) return;
  int e = idx >> 7, c = idx & 127;
  float s = xji[idx];
#pragma unroll
  for (int b = 0; b < 8; ++b) s += agg[e * 8 + b] * upW[b * 128 + c];
  xF[idx] = s;
  xH[idx] = (_Float16)s;
}

// atom_out[a] += t[a,:] @ Wf[:,0]
__global__ void k_out_final(const float* __restrict__ tF, const float* __restrict__ Wf,
                            float* __restrict__ aout, int NA) {
  int a = blockIdx.x * blockDim.x + threadIdx.x;
  if (a >= NA) return;
  float s = 0.f;
  for (int c = 0; c < 128; ++c) s += tF[(size_t)a * 128 + c] * Wf[c];
  aout[a] += s;
}

__global__ void k_seg_geom(const float* __restrict__ aout, const int* __restrict__ geom,
                           float* __restrict__ pg, int NA, int G) {
  int idx = blockIdx.x * blockDim.x + threadIdx.x;
  if (idx >= 3 * NA) return;
  int k = idx / NA, a = idx % NA;
  atomicAdd(&pg[k * G + geom[a]], aout[idx]);
}

__global__ void k_adiabatic(const float* __restrict__ pg, float* __restrict__ out, int G) {
  int g = blockIdx.x * blockDim.x + threadIdx.x;
  if (g >= G) return;
  float d0 = pg[g], d1 = pg[G + g], lam = pg[2 * G + g];
  float avg = 0.5f * (d0 + d1);
  float dd = d0 - d1;
  float gap = sqrtf(0.25f * dd * dd + lam * lam + 1e-12f);
  out[2 * g]     = avg - gap;
  out[2 * g + 1] = avg + gap;
}

// ---------------------------------------------------------------------------
// Host orchestration
// ---------------------------------------------------------------------------
extern "C" void kernel_launch(void* const* d_in, const int* in_sizes, int n_in,
                              void* d_out, int out_size, void* d_ws, size_t ws_size,
                              hipStream_t stream) {
  (void)n_in; (void)ws_size;
  const float* xyz         = (const float*)d_in[0];
  const float* emb_table   = (const float*)d_in[1];
  const float* emb_rbf_W   = (const float*)d_in[2];
  const float* emb_rbf_b   = (const float*)d_in[3];
  const float* emb_dense_W = (const float*)d_in[4];
  const float* emb_dense_b = (const float*)d_in[5];
  const float* int_ji_W    = (const float*)d_in[6];
  const float* int_ji_b    = (const float*)d_in[7];
  const float* int_kj_W    = (const float*)d_in[8];
  const float* int_kj_b    = (const float*)d_in[9];
  const float* int_rbf_W   = (const float*)d_in[10];
  const float* int_bilin_W = (const float*)d_in[11];
  const float* int_up_W    = (const float*)d_in[12];
  const float* int_res_W1  = (const float*)d_in[13];
  const float* int_res_b1  = (const float*)d_in[14];
  const float* int_res_W2  = (const float*)d_in[15];
  const float* int_res_b2  = (const float*)d_in[16];
  const float* int_skip_W  = (const float*)d_in[17];
  const float* int_skip_b  = (const float*)d_in[18];
  const float* out_rbf_W   = (const float*)d_in[19];
  const float* out_dense_W = (const float*)d_in[20];
  const float* out_dense_b = (const float*)d_in[21];
  const float* out_final_W = (const float*)d_in[22];
  const int* z    = (const int*)d_in[23];
  const int* nbr  = (const int*)d_in[24];
  const int* ang  = (const int*)d_in[25];
  const int* kj   = (const int*)d_in[26];
  const int* ji   = (const int*)d_in[27];
  const int* geom = (const int*)d_in[28];

  const int E  = in_sizes[24] / 2;
  const int T  = in_sizes[25] / 3;
  const int NA = in_sizes[23];
  const int G  = out_size / 2;

  // bump allocator over workspace (256B aligned)
  size_t off = 0;
  auto alloc = [&](size_t bytes) -> char* {
    size_t a = (off + 255) & ~(size_t)255;
    off = a + bytes;
    return (char*)d_ws + a;
  };

  float*     dEdge  = (float*)alloc((size_t)E * 4);
  float*     erbf   = (float*)alloc((size_t)E * 6 * 4);
  float*     sbf    = (float*)alloc((size_t)T * 42 * 4);
  _Float16*  Aemb   = (_Float16*)alloc((size_t)E * 384 * 2);
  float*     mF     = (float*)alloc((size_t)E * 128 * 4);
  _Float16*  mH     = (_Float16*)alloc((size_t)E * 128 * 2);
  float*     xji    = (float*)alloc((size_t)E * 128 * 4);
  _Float16*  xkjH   = (_Float16*)alloc((size_t)E * 128 * 2);
  float*     xF     = (float*)alloc((size_t)E * 128 * 4);
  _Float16*  xH     = (_Float16*)alloc((size_t)E * 128 * 2);
  _Float16*  uH     = (_Float16*)alloc((size_t)E * 128 * 2);
  float*     mulbuf = (float*)alloc((size_t)E * 128 * 4);
  float*     agg    = (float*)alloc((size_t)E * 8 * 4);
  float*     tAf    = (float*)alloc((size_t)NA * 128 * 4);
  _Float16*  tAh    = (_Float16*)alloc((size_t)NA * 128 * 2);
  float*     tBf    = (float*)alloc((size_t)NA * 128 * 4);
  _Float16*  tBh    = (_Float16*)alloc((size_t)NA * 128 * 2);
  float*     aout   = (float*)alloc((size_t)3 * NA * 4);
  float*     pgeom  = (float*)alloc((size_t)3 * G * 4);

  _Float16* embWt = (_Float16*)alloc((size_t)384 * 128 * 2);
  _Float16 *jiWt[4], *kjWt[4], *skWt[4], *r1Wt[4][3], *r2Wt[4][3], *blWt[4];
  for (int L = 0; L < 4; ++L) {
    jiWt[L] = (_Float16*)alloc(128 * 128 * 2);
    kjWt[L] = (_Float16*)alloc(128 * 128 * 2);
    skWt[L] = (_Float16*)alloc(128 * 128 * 2);
    for (int r = 0; r < 3; ++r) {
      r1Wt[L][r] = (_Float16*)alloc(128 * 128 * 2);
      r2Wt[L][r] = (_Float16*)alloc(128 * 128 * 2);
    }
    blWt[L] = (_Float16*)alloc(336 * 128 * 2);
  }
  _Float16* odWt[3][5][3];
  for (int k = 0; k < 3; ++k)
    for (int i = 0; i < 5; ++i)
      for (int j = 0; j < 3; ++j)
        odWt[k][i][j] = (_Float16*)alloc(128 * 128 * 2);

  auto wtr = [&](const float* W, _Float16* Wt, int K, int N) {
    int n = K * N;
    k_w_tr<<<(n + 255) / 256, 256, 0, stream>>>(W, Wt, K, N);
  };
  auto gemm = [&](const _Float16* A, const _Float16* Wt, const float* bias,
                  const float* mul, const float* resid, float* oF, _Float16* oH,
                  int M, int K, int act) {
    k_gemm128<<<dim3((M + 127) / 128), 256, 0, stream>>>(A, Wt, bias, mul, resid,
                                                         oF, oH, M, K, act);
  };

  // ---- weight conversion (f32 -> f16 transposed) ----
  wtr(emb_dense_W, embWt, 384, 128);
  for (int L = 0; L < 4; ++L) {
    wtr(int_ji_W   + (size_t)L * 16384, jiWt[L], 128, 128);
    wtr(int_kj_W   + (size_t)L * 16384, kjWt[L], 128, 128);
    wtr(int_skip_W + (size_t)L * 16384, skWt[L], 128, 128);
    for (int r = 0; r < 3; ++r) {
      wtr(int_res_W1 + (size_t)(L * 3 + r) * 16384, r1Wt[L][r], 128, 128);
      wtr(int_res_W2 + (size_t)(L * 3 + r) * 16384, r2Wt[L][r], 128, 128);
    }
    k_bilin_tr<<<(42 * 128 * 8 + 255) / 256, 256, 0, stream>>>(
        int_bilin_W + (size_t)L * 43008, blWt[L]);
  }
  for (int k = 0; k < 3; ++k)
    for (int i = 0; i < 5; ++i)
      for (int j = 0; j < 3; ++j)
        wtr(out_dense_W + (size_t)((k * 5 + i) * 3 + j) * 16384, odWt[k][i][j], 128, 128);

  // ---- geometry ----
  k_edge_geom<<<(E + 255) / 256, 256, 0, stream>>>(xyz, nbr, dEdge, erbf, E);
  k_sbf<<<(T + 255) / 256, 256, 0, stream>>>(xyz, ang, kj, dEdge, sbf, T);

  // ---- embedding ----
  k_build_emb<<<(E * 128 + 255) / 256, 256, 0, stream>>>(nbr, z, emb_table, erbf,
                                                         emb_rbf_W, emb_rbf_b, Aemb, E);
  gemm(Aemb, embWt, emb_dense_b, nullptr, nullptr, mF, mH, E, 384, 1);

  hipMemsetAsync(aout, 0, (size_t)3 * NA * 4, stream);
  hipMemsetAsync(pgeom, 0, (size_t)3 * G * 4, stream);

  // ---- output block ----
  auto outblk = [&](int k, int i) {
    size_t ki = (size_t)(k * 5 + i);
    hipMemsetAsync(tAf, 0, (size_t)NA * 128 * 4, stream);
    k_out_edge<<<(E * 128 + 255) / 256, 256, 0, stream>>>(
        erbf, out_rbf_W + ki * 768, mF, nbr, tAf, E);
    k_cast_h<<<(NA * 128 + 255) / 256, 256, 0, stream>>>(tAf, tAh, NA * 128);
    const float* bB = out_dense_b + ki * 3 * 128;
    gemm(tAh, odWt[k][i][0], bB,           nullptr, nullptr, tBf, tBh, NA, 128, 1);
    gemm(tBh, odWt[k][i][1], bB + 128,     nullptr, nullptr, tAf, tAh, NA, 128, 1);
    gemm(tAh, odWt[k][i][2], bB + 256,     nullptr, nullptr, tBf, tBh, NA, 128, 1);
    k_out_final<<<(NA + 255) / 256, 256, 0, stream>>>(
        tBf, out_final_W + ki * 128, aout + (size_t)k * NA, NA);
  };
  for (int k = 0; k < 3; ++k) outblk(k, 0);

  // ---- residual helper ----
  auto resblk = [&](float* vF, _Float16* vH, const _Float16* W1t, const float* b1,
                    const _Float16* W2t, const float* b2) {
    gemm(vH, W1t, b1, nullptr, nullptr, nullptr, uH, E, 128, 1);
    gemm(uH, W2t, b2, nullptr, vF, vF, vH, E, 128, 1);   // v + silu(u@W2+b2)
  };

  // ---- interaction layers ----
  for (int L = 0; L < 4; ++L) {
    k_rbfproj<<<(E * 128 + 255) / 256, 256, 0, stream>>>(
        erbf, int_rbf_W + (size_t)L * 768, mulbuf, E);
    gemm(mH, jiWt[L], int_ji_b + (size_t)L * 128, nullptr, nullptr, xji, nullptr, E, 128, 1);
    gemm(mH, kjWt[L], int_kj_b + (size_t)L * 128, mulbuf, nullptr, nullptr, xkjH, E, 128, 1);
    hipMemsetAsync(agg, 0, (size_t)E * 8 * 4, stream);
    k_bilin<<<dim3((T + 127) / 128), 256, 0, stream>>>(xkjH, blWt[L], sbf, kj, ji, agg, T);
    k_up<<<(E * 128 + 255) / 256, 256, 0, stream>>>(
        xji, agg, int_up_W + (size_t)L * 1024, xF, xH, E);
    resblk(xF, xH, r1Wt[L][0], int_res_b1 + (size_t)(L * 3) * 128,
                  r2Wt[L][0], int_res_b2 + (size_t)(L * 3) * 128);
    // m = silu(x @ skip + b) + m
    gemm(xH, skWt[L], int_skip_b + (size_t)L * 128, nullptr, mF, mF, mH, E, 128, 1);
    resblk(mF, mH, r1Wt[L][1], int_res_b1 + (size_t)(L * 3 + 1) * 128,
                  r2Wt[L][1], int_res_b2 + (size_t)(L * 3 + 1) * 128);
    resblk(mF, mH, r1Wt[L][2], int_res_b1 + (size_t)(L * 3 + 2) * 128,
                  r2Wt[L][2], int_res_b2 + (size_t)(L * 3 + 2) * 128);
    for (int k = 0; k < 3; ++k) outblk(k, L + 1);
  }

  // ---- per-geometry reduction + adiabatic combine ----
  k_seg_geom<<<(3 * NA + 255) / 256, 256, 0, stream>>>(aout, geom, pgeom, NA, G);
  k_adiabatic<<<(G + 255) / 256, 256, 0, stream>>>(pgeom, (float*)d_out, G);
}
